// SpotEncoder_85555748536633
// MI455X (gfx1250) — matrix operand order repaired and checked by reading, not compiled, not run on hardware
//
#include <hip/hip_runtime.h>

#define D 128
#define EPSV 1e-5f
#define EPW 4  // edges per wave in scatter

typedef __attribute__((ext_vector_type(16))) __bf16 v16bf;
typedef __attribute__((ext_vector_type(8)))  float  v8f;

union Frag16 { v16bf v; uint4 q[2]; };

__device__ __forceinline__ unsigned short f2bf(float f) {
  // round-to-nearest-even f32 -> bf16
  unsigned int u = __float_as_uint(f);
  u += 0x7FFFu + ((u >> 16) & 1u);
  return (unsigned short)(u >> 16);
}

// ---------------- degree / norm ----------------
__global__ void k_fill1(float* p, int n) {
  int i = blockIdx.x * blockDim.x + threadIdx.x;
  if (i < n) p[i] = 1.0f;  // self-loop contributes 1 to every degree
}

__global__ void k_deg(const int* __restrict__ ei, int E, float* deg) {
  int i = blockIdx.x * blockDim.x + threadIdx.x;
  if (i < E) unsafeAtomicAdd(&deg[ei[E + i]], 1.0f);  // dst row of edge_index
}

__global__ void k_rsqrt(float* p, int n) {
  int i = blockIdx.x * blockDim.x + threadIdx.x;
  if (i < n) p[i] = rsqrtf(p[i]);  // deg >= 1 always (self-loops)
}

// ---------------- W -> bf16 transpose: Wt[n][k] = bf16(W[k][n]) ----------------
__global__ void k_wcvt(const float* __restrict__ W, unsigned short* __restrict__ Wt) {
  int t = blockIdx.x * blockDim.x + threadIdx.x;  // 16384 total
  int n = t >> 7, k = t & 127;
  Wt[n * D + k] = f2bf(W[k * D + n]);
}

// ---------------- fused LayerNorm -> ReLU -> (bf16) GEMM via WMMA ----------------
// grid: ceil(N/16) blocks of 256 threads (8 waves). Wave w computes cols [16w,16w+16).
__global__ __launch_bounds__(256)
void k_ln_gemm(const float* __restrict__ h, const float* __restrict__ g,
               const float* __restrict__ bt, const unsigned short* __restrict__ Wt,
               float* __restrict__ out, int N) {
  __shared__ float h_s[16 * D];                          // 8 KB staging
  __shared__ __align__(16) unsigned short a_s[16][136];  // padded bf16 A tile (4.25 KB)

  const int t = threadIdx.x;
  const int tile0 = blockIdx.x * 16;
  const bool full = (tile0 + 16 <= N);  // uniform across block

  // stage 16 rows (8 floats / thread, two float4s)
  {
    int f = t * 8;
    const float* src = h + (size_t)tile0 * D + f;
    if (full) {
      *(float4*)&h_s[f]     = *(const float4*)(src);
      *(float4*)&h_s[f + 4] = *(const float4*)(src + 4);
    } else {
      int row = f >> 7;
      float4 v0 = make_float4(0.f, 0.f, 0.f, 0.f), v1 = v0;
      if (tile0 + row < N) { v0 = *(const float4*)(src); v1 = *(const float4*)(src + 4); }
      *(float4*)&h_s[f]     = v0;
      *(float4*)&h_s[f + 4] = v1;
    }
  }
  __syncthreads();

  // per-row LayerNorm stats: 16 lanes per row, xor-shuffle reduction stays in 16-lane groups
  const int row = t >> 4;
  const int c0 = t & 15;
  float sum = 0.f, ss = 0.f;
#pragma unroll
  for (int j = 0; j < 8; j++) {
    float v = h_s[row * D + c0 + 16 * j];
    sum += v; ss += v * v;
  }
#pragma unroll
  for (int m = 1; m < 16; m <<= 1) {
    sum += __shfl_xor(sum, m, 32);
    ss  += __shfl_xor(ss,  m, 32);
  }
  const float mu = sum * (1.0f / D);
  const float var = ss * (1.0f / D) - mu * mu;
  const float rstd = rsqrtf(var + EPSV);

  // apply LN + affine + ReLU, round to bf16 into A tile
#pragma unroll
  for (int j = 0; j < 8; j++) {
    int c = c0 + 16 * j;
    float v = (h_s[row * D + c] - mu) * rstd * g[c] + bt[c];
    a_s[row][c] = f2bf(fmaxf(v, 0.0f));
  }
  __syncthreads();

  // WMMA: D(16x16) += A(16x32) x B(32x16) over 4 k-tiles
  const int lane = t & 31;
  const int wv = t >> 5;        // N-tile id 0..7
  const int half = lane >> 4;   // 0 / 1
  const int l15 = lane & 15;
  const unsigned short* wrow = Wt + (size_t)(wv * 16 + l15) * D;  // column (wv*16+l15) of W

  v8f acc = {0.f, 0.f, 0.f, 0.f, 0.f, 0.f, 0.f, 0.f};
#pragma unroll
  for (int kt = 0; kt < 4; kt++) {
    Frag16 a, b;
    // A layout (16-bit 16x32): lanes 0-15 -> K {0-7,16-23}; lanes 16-31 -> K {8-15,24-31}
    const int kbA = kt * 32 + half * 8;
    a.q[0] = *(const uint4*)&a_s[l15][kbA];
    a.q[1] = *(const uint4*)&a_s[l15][kbA + 16];
    // B layout (16-bit 32x16): lanes 0-15 -> K {0-7,16-23}; lanes 16-31 -> K {8-15,24-31}
    const int kbB = kt * 32 + half * 8;
    b.q[0] = *(const uint4*)(wrow + kbB);
    b.q[1] = *(const uint4*)(wrow + kbB + 16);
    acc = __builtin_amdgcn_wmma_f32_16x16x32_bf16(false, a.v, false, b.v,
                                                  (short)0, acc, false, false);
  }

  // D layout: lanes 0-15 rows 0-7, lanes 16-31 rows 8-15; col = lane&15 within tile
  const int col = wv * 16 + l15;
  float* orow = out + (size_t)(tile0 + half * 8) * D + col;
  if (full) {  // uniform branch -> unguarded store fast path
#pragma unroll
    for (int j = 0; j < 8; j++) orow[(size_t)j * D] = acc[j];
  } else {
#pragma unroll
    for (int j = 0; j < 8; j++)
      if (tile0 + half * 8 + j < N) orow[(size_t)j * D] = acc[j];
  }
}

// ---------------- out init: bias + optional residual + self-loop term ----------------
__global__ void k_init_out(float* __restrict__ out, const float* __restrict__ bias,
                           const float* __restrict__ res, const float* __restrict__ dinv,
                           const float* __restrict__ hn, int total) {
  int i = blockIdx.x * blockDim.x + threadIdx.x;
  if (i >= total) return;
  int node = i >> 7, d = i & 127;
  float di = dinv[node];
  float v = bias[d] + di * di * hn[i];
  if (res) v += res[i];
  out[i] = v;
}

// ---------------- edge scatter: out[dst] += dinv[src]*dinv[dst] * hn[src] ----------------
__global__ __launch_bounds__(256)
void k_scatter(const int* __restrict__ ei, int E, const float* __restrict__ dinv,
               const float* __restrict__ hn, float* __restrict__ out) {
  const int lane = threadIdx.x & 31;
  const int wave = blockIdx.x * 8 + (threadIdx.x >> 5);
  const int e0 = wave * EPW;
#pragma unroll
  for (int k = 0; k < EPW; k++) {
    int e = e0 + k;
    if (e >= E) return;
    int src = ei[e];
    int dst = ei[E + e];
    if (k + 1 < EPW && e + 1 < E) {
      int s2 = ei[e + 1];
      __builtin_prefetch(&hn[(size_t)s2 * D + lane * 4], 0, 0);
    }
    float nrm = dinv[src] * dinv[dst];
    float4 v = *(const float4*)&hn[(size_t)src * D + lane * 4];
    float* o = out + (size_t)dst * D + lane * 4;
    unsafeAtomicAdd(o + 0, nrm * v.x);
    unsafeAtomicAdd(o + 1, nrm * v.y);
    unsafeAtomicAdd(o + 2, nrm * v.z);
    unsafeAtomicAdd(o + 3, nrm * v.w);
  }
}

extern "C" void kernel_launch(void* const* d_in, const int* in_sizes, int n_in,
                              void* d_out, int out_size, void* d_ws, size_t ws_size,
                              hipStream_t stream) {
  const float* x   = (const float*)d_in[0];
  const int*   ei  = (const int*)d_in[1];
  const float* lng[3]  = {(const float*)d_in[2], (const float*)d_in[6],  (const float*)d_in[10]};
  const float* lnb[3]  = {(const float*)d_in[3], (const float*)d_in[7],  (const float*)d_in[11]};
  const float* W[3]    = {(const float*)d_in[4], (const float*)d_in[8],  (const float*)d_in[12]};
  const float* bias[3] = {(const float*)d_in[5], (const float*)d_in[9],  (const float*)d_in[13]};
  const int Nn = in_sizes[0] / D;
  const int E  = in_sizes[1] / 2;
  float* out = (float*)d_out;

  // workspace: dinv | Wt(bf16) | hn | x1   (~51.5 MB)
  char* ws = (char*)d_ws;
  size_t off = 0;
  float* dinv = (float*)ws;
  off += ((size_t)Nn * 4 + 255) & ~(size_t)255;
  unsigned short* Wt = (unsigned short*)(ws + off);
  off += (size_t)D * D * 2;
  float* hn = (float*)(ws + off);
  off += (size_t)Nn * D * 4;
  float* x1 = (float*)(ws + off);

  const int total = Nn * D;
  const dim3 b256(256);
  const int gemmBlocks = (Nn + 15) / 16;
  const int scatBlocks = (E + 8 * EPW - 1) / (8 * EPW);
  const int elemBlocks = (total + 255) / 256;

  // GCN normalization
  k_fill1<<<(Nn + 255) / 256, b256, 0, stream>>>(dinv, Nn);
  k_deg<<<(E + 255) / 256, b256, 0, stream>>>(ei, E, dinv);
  k_rsqrt<<<(Nn + 255) / 256, b256, 0, stream>>>(dinv, Nn);

  // layer 0: x -> x1, residual x
  k_wcvt<<<64, b256, 0, stream>>>(W[0], Wt);
  k_ln_gemm<<<gemmBlocks, b256, 0, stream>>>(x, lng[0], lnb[0], Wt, hn, Nn);
  k_init_out<<<elemBlocks, b256, 0, stream>>>(x1, bias[0], x, dinv, hn, total);
  k_scatter<<<scatBlocks, b256, 0, stream>>>(ei, E, dinv, hn, x1);

  // layer 1: x1 -> h1 (stored in d_out), no residual
  k_wcvt<<<64, b256, 0, stream>>>(W[1], Wt);
  k_ln_gemm<<<gemmBlocks, b256, 0, stream>>>(x1, lng[1], lnb[1], Wt, hn, Nn);
  k_init_out<<<elemBlocks, b256, 0, stream>>>(out, bias[1], nullptr, dinv, hn, total);
  k_scatter<<<scatBlocks, b256, 0, stream>>>(ei, E, dinv, hn, out);

  // layer 2: h1 (d_out) -> d_out, residual x1
  k_wcvt<<<64, b256, 0, stream>>>(W[2], Wt);
  k_ln_gemm<<<gemmBlocks, b256, 0, stream>>>(out, lng[2], lnb[2], Wt, hn, Nn);
  k_init_out<<<elemBlocks, b256, 0, stream>>>(out, bias[2], x1, dinv, hn, total);
  k_scatter<<<scatBlocks, b256, 0, stream>>>(ei, E, dinv, hn, out);
}
